// MultiHeadAttention_4234837754413
// MI455X (gfx1250) — matrix operand order repaired and checked by reading, not compiled, not run on hardware
//
#include <hip/hip_runtime.h>
#include <hip/hip_bf16.h>

#define B_  4
#define S_  2048
#define D_  1024
#define H_  16
#define DH_ 64
#define M_  (B_*S_)

typedef __attribute__((ext_vector_type(16))) __bf16 v16bf;
typedef __attribute__((ext_vector_type(8)))  __bf16 v8bf;
typedef __attribute__((ext_vector_type(4)))  __bf16 v4bf;
typedef __attribute__((ext_vector_type(8)))  float  v8f;
typedef __attribute__((ext_vector_type(4)))  unsigned int u32x4;
typedef __attribute__((ext_vector_type(8)))  unsigned int u32x8;

__device__ __forceinline__ v8f wmma_bf16(v16bf a, v16bf b, v8f c) {
  return __builtin_amdgcn_wmma_f32_16x16x32_bf16(
      /*neg_a=*/false, a, /*neg_b=*/false, b,
      /*c_mod=*/(short)0, c, /*reuse_a=*/false, /*reuse_b=*/false);
}

// Build a 16x32 bf16 A-fragment from a row-major row pointer.
// Layout: lanes 0-15 (row = lane)  hold K = {0..15}
//         lanes 16-31 (row=lane-16) hold K = {16..31}
__device__ __forceinline__ v16bf make_afrag(const __bf16* rowp, int lane) {
  const int kb = (lane & 16) ? 16 : 0;
  v8bf lo = *(const v8bf*)(rowp + kb);
  v8bf hi = *(const v8bf*)(rowp + kb + 8);
  v16bf r;
#pragma unroll
  for (int i = 0; i < 8; ++i) { r[i] = lo[i]; r[i + 8] = hi[i]; }
  return r;
}

// ---- Tensor Data Mover: 2D tile (bf16) global -> LDS, per ISA 08_async_tensor §8 ----
// D# group0: count=1 | lds_addr | global_addr[56:0] | type=2
// D# group1: data_size=1(2B) | tensor_dim0/1 | tile_dim0/1 | tensor_dim0_stride
__device__ __forceinline__ void tdm_load_2d_bf16(
    unsigned lds_byte_addr, const void* gptr,
    unsigned tensor_d0, unsigned tensor_d1,
    unsigned tile_d0, unsigned tile_d1, unsigned stride0_elems) {
  unsigned long long ga = (unsigned long long)(size_t)gptr;
  u32x4 g0;
  g0[0] = 1u;                                    // count=1, no gather, user desc
  g0[1] = lds_byte_addr;                         // lds_addr (bytes)
  g0[2] = (unsigned)(ga & 0xffffffffu);          // global_addr[31:0]
  g0[3] = (unsigned)((ga >> 32) & 0x01ffffffu) | (2u << 30);  // addr[56:32] | type=2
  u32x8 g1;
  g1[0] = 0x00010000u;                           // wg_mask=0, data_size=1 (2 bytes)
  g1[1] = (tensor_d0 & 0xffffu) << 16;           // tensor_dim0[15:0] at bits 63:48
  g1[2] = (tensor_d0 >> 16) | ((tensor_d1 & 0xffffu) << 16);
  g1[3] = (tensor_d1 >> 16) | (tile_d0 << 16);   // tile_dim0 at bits 127:112
  g1[4] = tile_d1;                               // tile_dim1; tile_dim2=0
  g1[5] = stride0_elems;                         // tensor_dim0_stride[31:0]
  g1[6] = 0u;                                    // stride0[47:32]=0, stride1 lo=0
  g1[7] = 0u;
  asm volatile("tensor_load_to_lds %0, %1" :: "s"(g0), "s"(g1) : "memory");
}

// ---------------- f32 -> bf16 conversion ----------------
__global__ void cvt_f32_bf16_kernel(const float* __restrict__ src,
                                    __bf16* __restrict__ dst, int n4) {
  int i = blockIdx.x * blockDim.x + threadIdx.x;
  if (i >= n4) return;
  float4 f = ((const float4*)src)[i];
  v4bf o = { (__bf16)f.x, (__bf16)f.y, (__bf16)f.z, (__bf16)f.w };
  ((v4bf*)dst)[i] = o;
}

// ---------------- tiled bf16 GEMM, Y = X(MxK) @ W(KxN) + bias ----------------
// Double-buffered LDS; A tiles arrive via TDM (tensor_load_to_lds), overlapped
// with WMMA on the other buffer. MODE 0: bf16 head-split out; MODE 1: f32 out.
template <int MODE>
__global__ __launch_bounds__(256) void gemm_bf16_kernel(
    const __bf16* __restrict__ X, const __bf16* __restrict__ W,
    const float* __restrict__ bias, void* __restrict__ out) {
  __shared__ alignas(32) __bf16 As[2][128 * 32];   // [m][k]
  __shared__ alignas(32) __bf16 Bts[2][128 * 32];  // [n][k] (pre-transposed)

  const int tid  = threadIdx.x;
  const int lane = tid & 31;
  const int wave = tid >> 5;   // 0..7
  const int wm   = wave >> 1;  // 0..3  (32-row slabs)
  const int wn   = wave & 1;   // 0..1  (64-col slabs)
  const int m0   = blockIdx.y * 128;
  const int n0   = blockIdx.x * 128;

  const v8f vzero = {0.f, 0.f, 0.f, 0.f, 0.f, 0.f, 0.f, 0.f};
  v8f acc[2][4];
#pragma unroll
  for (int i = 0; i < 2; ++i)
#pragma unroll
    for (int j = 0; j < 4; ++j) acc[i][j] = vzero;

  // B tile 32x128 -> LDS transposed [n][k]
  auto stageB = [&](int k0, int buf) {
#pragma unroll
    for (int it = 0; it < 2; ++it) {
      int c = tid + it * 256;
      int kr = c >> 4, no = (c & 15) * 8;
      v8bf w = *(const v8bf*)&W[(size_t)(k0 + kr) * D_ + n0 + no];
#pragma unroll
      for (int j = 0; j < 8; ++j) Bts[buf][(no + j) * 32 + kr] = w[j];
    }
  };

  // prologue: fill buffer 0
  if (wave == 0)
    tdm_load_2d_bf16((unsigned)(size_t)&As[0][0], &X[(size_t)m0 * D_],
                     D_, M_, 32, 128, D_);
  stageB(0, 0);
  if (wave == 0) __builtin_amdgcn_s_wait_tensorcnt(0);
  __syncthreads();

  int cur = 0;
  const int NK = D_ / 32;
  for (int kt = 0; kt < NK; ++kt) {
    const int nxt = cur ^ 1;
    if (kt + 1 < NK) {  // kick off next tile while computing this one
      if (wave == 0)
        tdm_load_2d_bf16((unsigned)(size_t)&As[nxt][0],
                         &X[(size_t)m0 * D_ + (kt + 1) * 32],
                         D_, M_, 32, 128, D_);
      stageB((kt + 1) * 32, nxt);
    }

    v16bf af[2];
#pragma unroll
    for (int i = 0; i < 2; ++i) {
      int row = wm * 32 + i * 16 + (lane & 15);
      af[i] = make_afrag(&As[cur][row * 32], lane);
    }
#pragma unroll
    for (int j = 0; j < 4; ++j) {
      int n = wn * 64 + j * 16 + (lane & 15);
      v16bf bfg = *(const v16bf*)&Bts[cur][n * 32 + ((lane & 16) ? 16 : 0)];
#pragma unroll
      for (int i = 0; i < 2; ++i) acc[i][j] = wmma_bf16(af[i], bfg, acc[i][j]);
    }

    if (wave == 0) __builtin_amdgcn_s_wait_tensorcnt(0);
    __syncthreads();
    cur = nxt;
  }

  // epilogue: C/D layout — lane holds col n = lane&15 (+16*frag), VGPR r -> row r + 8*(lane>=16)
#pragma unroll
  for (int i = 0; i < 2; ++i) {
#pragma unroll
    for (int j = 0; j < 4; ++j) {
      const int n = n0 + wn * 64 + j * 16 + (lane & 15);
      const float bv = bias[n];
#pragma unroll
      for (int r = 0; r < 8; ++r) {
        const int m = m0 + wm * 32 + i * 16 + r + ((lane & 16) ? 8 : 0);
        const float val = acc[i][j][r] + bv;
        if (MODE == 0) {
          const int b = m >> 11, s = m & (S_ - 1);
          const int h = n >> 6, dh = n & 63;
          ((__bf16*)out)[(((size_t)b * H_ + h) * S_ + s) * DH_ + dh] = (__bf16)val;
        } else {
          ((float*)out)[(size_t)m * D_ + n] = val;
        }
      }
    }
  }
}

// ---------------- flash attention over heads ----------------
// grid: (S/64, B*H), 128 threads = 4 waves, each wave owns 16 q-rows.
// K chunk staged with global_load_async_to_lds_b128 (ASYNCcnt).
__global__ __launch_bounds__(128) void attn_kernel(
    const __bf16* __restrict__ Qh, const __bf16* __restrict__ Kh,
    const __bf16* __restrict__ Vh, __bf16* __restrict__ ctx) {
  __shared__ alignas(32) __bf16 Ks[64 * 64];       // [key][dh]  (B-op for QK^T)
  __shared__ alignas(32) __bf16 Vt[64 * 64];       // [dh][key]  (B-op for P@V)
  __shared__ alignas(32) __bf16 Ps[4 * 16 * 64];   // per-wave P staging

  const int tid = threadIdx.x, lane = tid & 31, wave = tid >> 5;
  const int bh = blockIdx.y;
  const int b = bh >> 4, h = bh & 15;
  const int q0 = blockIdx.x * 64;
  const int qw = q0 + wave * 16;
  const size_t hbase = (size_t)bh * S_ * DH_;

  // Q A-fragments (DH=64 -> two K-steps of 32), kept in registers
  v16bf qf[2];
  {
    const __bf16* qp = Qh + hbase + (size_t)(qw + (lane & 15)) * DH_;
    qf[0] = make_afrag(qp, lane);
    qf[1] = make_afrag(qp + 32, lane);
  }

  const v8f vzero = {0.f, 0.f, 0.f, 0.f, 0.f, 0.f, 0.f, 0.f};
  v8f co[4];
#pragma unroll
  for (int f = 0; f < 4; ++f) co[f] = vzero;
  float vm[8], l[8];
#pragma unroll
  for (int r = 0; r < 8; ++r) { vm[r] = -3.0e38f; l[r] = 0.f; }

  for (int kc = 0; kc < S_; kc += 64) {
    __syncthreads();
    // stage K via async copy to LDS; V via transposed ds stores
#pragma unroll
    for (int it = 0; it < 4; ++it) {
      int c = tid + it * 128;
      int row = c >> 3, off = (c & 7) * 8;
      unsigned lk = (unsigned)(size_t)&Ks[row * 64 + off];
      unsigned long long gk =
          (unsigned long long)(size_t)&Kh[hbase + (size_t)(kc + row) * DH_ + off];
      asm volatile("global_load_async_to_lds_b128 %0, %1, off"
                   :: "v"(lk), "v"(gk) : "memory");
      v8bf vv = *(const v8bf*)&Vh[hbase + (size_t)(kc + row) * DH_ + off];
#pragma unroll
      for (int j = 0; j < 8; ++j) Vt[(off + j) * 64 + row] = vv[j];
    }
    asm volatile("s_wait_asynccnt 0" ::: "memory");
    __syncthreads();

    // scores 16x64 = Q @ K^T, scaled by 1/sqrt(H) = 0.25 (reference quirk)
    v8f sc[4];
#pragma unroll
    for (int f = 0; f < 4; ++f) {
      sc[f] = vzero;
      const int n = f * 16 + (lane & 15);
#pragma unroll
      for (int kk = 0; kk < 2; ++kk) {
        v16bf bfg = *(const v16bf*)&Ks[n * 64 + kk * 32 + ((lane & 16) ? 16 : 0)];
        sc[f] = wmma_bf16(qf[kk], bfg, sc[f]);
      }
#pragma unroll
      for (int r = 0; r < 8; ++r) sc[f][r] *= 0.25f;
    }

    // online softmax: row stats live in the 16-lane half that owns the row
    float t[8];
#pragma unroll
    for (int r = 0; r < 8; ++r)
      t[r] = fmaxf(fmaxf(sc[0][r], sc[1][r]), fmaxf(sc[2][r], sc[3][r]));
#pragma unroll
    for (int mask = 1; mask <= 8; mask <<= 1)
#pragma unroll
      for (int r = 0; r < 8; ++r) t[r] = fmaxf(t[r], __shfl_xor(t[r], mask, 32));

    float corr[8];
#pragma unroll
    for (int r = 0; r < 8; ++r) {
      float mnew = fmaxf(vm[r], t[r]);
      corr[r] = __expf(vm[r] - mnew);
      vm[r] = mnew;
    }
    float ps[8];
#pragma unroll
    for (int r = 0; r < 8; ++r) ps[r] = 0.f;
#pragma unroll
    for (int f = 0; f < 4; ++f)
#pragma unroll
      for (int r = 0; r < 8; ++r) {
        float p = __expf(sc[f][r] - vm[r]);
        sc[f][r] = p;
        ps[r] += p;
      }
#pragma unroll
    for (int mask = 1; mask <= 8; mask <<= 1)
#pragma unroll
      for (int r = 0; r < 8; ++r) ps[r] += __shfl_xor(ps[r], mask, 32);
#pragma unroll
    for (int r = 0; r < 8; ++r) l[r] = l[r] * corr[r] + ps[r];
#pragma unroll
    for (int f = 0; f < 4; ++f)
#pragma unroll
      for (int r = 0; r < 8; ++r) co[f][r] *= corr[r];

    // C-layout -> A-layout via LDS (bf16)
    __bf16* pw = &Ps[wave * 16 * 64];
#pragma unroll
    for (int f = 0; f < 4; ++f)
#pragma unroll
      for (int r = 0; r < 8; ++r) {
        int m = r + ((lane & 16) ? 8 : 0);
        pw[m * 64 + f * 16 + (lane & 15)] = (__bf16)sc[f][r];
      }
    __syncthreads();

    // ctx += P @ V
    const __bf16* pr = &Ps[wave * 16 * 64 + (size_t)(lane & 15) * 64];
    v16bf pf[2];
    pf[0] = make_afrag(pr, lane);
    pf[1] = make_afrag(pr + 32, lane);
#pragma unroll
    for (int f = 0; f < 4; ++f) {
      const int n = f * 16 + (lane & 15);
#pragma unroll
      for (int kk = 0; kk < 2; ++kk) {
        v16bf bfg = *(const v16bf*)&Vt[n * 64 + kk * 32 + ((lane & 16) ? 16 : 0)];
        co[f] = wmma_bf16(pf[kk], bfg, co[f]);
      }
    }
  }

  // normalize and write ctx in concat-head layout [B,S,D]
#pragma unroll
  for (int f = 0; f < 4; ++f) {
    const int n = h * 64 + f * 16 + (lane & 15);
#pragma unroll
    for (int r = 0; r < 8; ++r) {
      const int m = qw + r + ((lane & 16) ? 8 : 0);
      ctx[((size_t)b * S_ + m) * D_ + n] = (__bf16)(co[f][r] / l[r]);
    }
  }
}

extern "C" void kernel_launch(void* const* d_in, const int* in_sizes, int n_in,
                              void* d_out, int out_size, void* d_ws, size_t ws_size,
                              hipStream_t stream) {
  (void)in_sizes; (void)n_in; (void)out_size; (void)ws_size;
  const float* q  = (const float*)d_in[0];
  const float* k  = (const float*)d_in[1];
  const float* v  = (const float*)d_in[2];
  const float* wq = (const float*)d_in[3];
  const float* bq = (const float*)d_in[4];
  const float* wk = (const float*)d_in[5];
  const float* bk = (const float*)d_in[6];
  const float* wv = (const float*)d_in[7];
  const float* bv = (const float*)d_in[8];
  const float* wo = (const float*)d_in[9];
  const float* bo = (const float*)d_in[10];

  char* ws = (char*)d_ws;
  size_t off = 0;
  auto alloc = [&](size_t bytes) -> void* {
    void* p = ws + off;
    off += (bytes + 255) & ~(size_t)255;
    return p;
  };
  const size_t xin_b = (size_t)M_ * D_ * sizeof(__bf16);
  const size_t w_b   = (size_t)D_ * D_ * sizeof(__bf16);
  __bf16* qb  = (__bf16*)alloc(xin_b);
  __bf16* kb  = (__bf16*)alloc(xin_b);
  __bf16* vb  = (__bf16*)alloc(xin_b);
  __bf16* wqb = (__bf16*)alloc(w_b);
  __bf16* wkb = (__bf16*)alloc(w_b);
  __bf16* wvb = (__bf16*)alloc(w_b);
  __bf16* wob = (__bf16*)alloc(w_b);
  __bf16* Qh  = (__bf16*)alloc(xin_b);   // [B,H,S,DH] bf16
  __bf16* Kh  = (__bf16*)alloc(xin_b);
  __bf16* Vh  = (__bf16*)alloc(xin_b);
  __bf16* ctxb = qb;  // qb dead after Q projection; reuse for ctx

  // 1) convert inputs + weights to bf16
  {
    int n4 = M_ * D_ / 4, blk = 256, g = (n4 + blk - 1) / blk;
    cvt_f32_bf16_kernel<<<g, blk, 0, stream>>>(q, qb, n4);
    cvt_f32_bf16_kernel<<<g, blk, 0, stream>>>(k, kb, n4);
    cvt_f32_bf16_kernel<<<g, blk, 0, stream>>>(v, vb, n4);
    int n4w = D_ * D_ / 4, gw = (n4w + blk - 1) / blk;
    cvt_f32_bf16_kernel<<<gw, blk, 0, stream>>>(wq, wqb, n4w);
    cvt_f32_bf16_kernel<<<gw, blk, 0, stream>>>(wk, wkb, n4w);
    cvt_f32_bf16_kernel<<<gw, blk, 0, stream>>>(wv, wvb, n4w);
    cvt_f32_bf16_kernel<<<gw, blk, 0, stream>>>(wo, wob, n4w);
  }

  // 2) Q/K/V projections -> head-split bf16
  dim3 ggrid(D_ / 128, M_ / 128);
  gemm_bf16_kernel<0><<<ggrid, 256, 0, stream>>>(qb, wqb, bq, (void*)Qh);
  gemm_bf16_kernel<0><<<ggrid, 256, 0, stream>>>(kb, wkb, bk, (void*)Kh);
  gemm_bf16_kernel<0><<<ggrid, 256, 0, stream>>>(vb, wvb, bv, (void*)Vh);

  // 3) flash attention -> ctx [B,S,D] bf16
  attn_kernel<<<dim3(S_ / 64, B_ * H_), 128, 0, stream>>>(Qh, Kh, Vh, ctxb);

  // 4) output projection -> f32 d_out
  gemm_bf16_kernel<1><<<ggrid, 256, 0, stream>>>(ctxb, wob, bo, d_out);
}
